// CrossAttention_52561809768894
// MI455X (gfx1250) — compile-verified, matrix operand order checked by reading, not run on hardware
//
#include <hip/hip_runtime.h>

#define HDIM 256
#define NSEQ 4096
#define NB 4

typedef __attribute__((ext_vector_type(16))) __bf16 v16bf;
typedef __attribute__((ext_vector_type(8)))  __bf16 v8bf;
typedef __attribute__((ext_vector_type(8)))  float  v8f;

// Build a 16x32 (A, lane=M) or 32x16 (B, lane=N) bf16 fragment from a row-major
// row pointer. Documented 16-bit A layout: lane group 0 holds K {0-7,16-23},
// group 1 holds K {8-15,24-31}. Two contiguous 16B loads per lane.
static __device__ __forceinline__ v16bf frag_from_bf(const __bf16* p, int hi) {
  v8bf a = *(const v8bf*)(p + hi * 8);
  v8bf b = *(const v8bf*)(p + 16 + hi * 8);
  v16bf r;
#pragma unroll
  for (int i = 0; i < 8; ++i) { r[i] = a[i]; r[i + 8] = b[i]; }
  return r;
}

// ---------------------------------------------------------------------------
// Kernel 0: one-shot fp32 -> bf16 conversion (x and the three W matrices).
// Done once so the GEMM K-loops never pay per-reuse v_cvt VALU cost.
// ---------------------------------------------------------------------------
__global__ __launch_bounds__(256) void cvt_bf16_kernel(
    const float* __restrict__ src, __bf16* __restrict__ dst, int n8)
{
  const int i = blockIdx.x * blockDim.x + threadIdx.x;
  if (i >= n8) return;
  v8f a = *(const v8f*)(src + (size_t)i * 8);
  alignas(16) __bf16 t[8];
#pragma unroll
  for (int v = 0; v < 8; ++v) t[v] = (__bf16)a[v];
  *(uint4*)(dst + (size_t)i * 8) = *(const uint4*)t;
}

// ---------------------------------------------------------------------------
// Kernel 1: QKV projection. y = x @ W^T + b (bf16 in, f32 accum, bf16 out).
// One 16x16 output tile per wave; Q,K stored row-major; V stored transposed
// (Vt[b][h][n]) so the P@V B-operand later loads contiguous key runs.
// ---------------------------------------------------------------------------
__global__ __launch_bounds__(256) void qkv_proj_kernel(
    const __bf16* __restrict__ xb, const __bf16* __restrict__ Wb,
    const float* __restrict__ bq, const float* __restrict__ bk,
    const float* __restrict__ bv,
    __bf16* __restrict__ Qb, __bf16* __restrict__ Kb, __bf16* __restrict__ Vt)
{
  const int lane = threadIdx.x & 31;
  const int l16  = lane & 15;
  const int hi   = lane >> 4;

  const int wid   = blockIdx.x * (blockDim.x >> 5) + (threadIdx.x >> 5);
  const int mtile = wid / 48;          // 1024 row tiles of 16
  const int r     = wid % 48;
  const int mat   = r >> 4;            // 0=Q 1=K 2=V
  const int ntile = r & 15;            // 16 col tiles of 16

  const float* bias = (mat == 0) ? bq : (mat == 1) ? bk : bv;

  const __bf16* xrow = xb + (size_t)(mtile * 16 + l16) * HDIM;   // A: lane = M row
  const __bf16* wrow = Wb + (size_t)mat * HDIM * HDIM            // B: lane = N row
                          + (size_t)(ntile * 16 + l16) * HDIM;   //    (B[k,n] = W[n,k])

  v8f acc = {};
#pragma unroll
  for (int k0 = 0; k0 < HDIM; k0 += 32) {
    v16bf a  = frag_from_bf(xrow + k0, hi);
    v16bf bf = frag_from_bf(wrow + k0, hi);
    acc = __builtin_amdgcn_wmma_f32_16x16x32_bf16(false, a, false, bf,
                                                  (short)0, acc, false, false);
  }

  const int   col = ntile * 16 + l16;      // D layout: lane holds column l16
  const float bb  = bias[col];

  if (mat < 2) {
    __bf16* dst = (mat == 0) ? Qb : Kb;
    const int row0 = mtile * 16 + 8 * hi;  // D: vgpr v -> row v + 8*(lane/16)
#pragma unroll
    for (int v = 0; v < 8; ++v)
      dst[(size_t)(row0 + v) * HDIM + col] = (__bf16)(acc[v] + bb);
  } else {
    const int g  = mtile * 16;
    const int b_ = g / NSEQ;
    const int n0 = (g % NSEQ) + 8 * hi;    // 8 consecutive sequence positions
    alignas(16) __bf16 tmp[8];
#pragma unroll
    for (int v = 0; v < 8; ++v) tmp[v] = (__bf16)(acc[v] + bb);
    *(uint4*)(Vt + (size_t)(b_ * HDIM + col) * NSEQ + n0) = *(const uint4*)tmp;
  }
}

// ---------------------------------------------------------------------------
// Kernel 2: flash attention. One 16-row query tile per wave; online softmax;
// 32 keys per step; full 16x256 f32 accumulator in registers (16 x v8f).
// ---------------------------------------------------------------------------
__global__ __launch_bounds__(128) void flash_attn_kernel(
    const __bf16* __restrict__ Qb, const __bf16* __restrict__ Kb,
    const __bf16* __restrict__ Vt, float* __restrict__ out)
{
  __shared__ __bf16 Pls[4][16][32];        // per-wave P-transpose staging (1KB/wave)

  const int lane = threadIdx.x & 31;
  const int l16  = lane & 15;
  const int hi   = lane >> 4;
  const int w    = threadIdx.x >> 5;

  const int qt = blockIdx.x * 4 + w;       // 0..1023 query tiles
  const int b  = qt >> 8;                  // batch (256 tiles per batch)
  const int r0 = qt * 16;                  // flat query row base

  // Q fragments held in registers for the whole key loop (8 x v16bf = 64 VGPRs)
  v16bf qa[8];
  const __bf16* qrow = Qb + (size_t)(r0 + l16) * HDIM;
#pragma unroll
  for (int t = 0; t < 8; ++t) qa[t] = frag_from_bf(qrow + t * 32, hi);

  v8f o[16];
#pragma unroll
  for (int ct = 0; ct < 16; ++ct) o[ct] = (v8f){};
  float mrow[8], lrow[8];
#pragma unroll
  for (int v = 0; v < 8; ++v) { mrow[v] = -1e30f; lrow[v] = 0.f; }

  const __bf16* Kbase = Kb + (size_t)b * NSEQ * HDIM;
  const __bf16* Vbase = Vt + (size_t)b * HDIM * NSEQ;

  for (int j0 = 0; j0 < NSEQ; j0 += 32) {
    // --- S = Q K^T for 32 keys: two 16x16 D tiles, K-depth 256 ---
    v8f s0 = {}, s1 = {};
    const __bf16* krow0 = Kbase + (size_t)(j0 + l16) * HDIM;   // B lane = key row
    const __bf16* krow1 = krow0 + 16 * HDIM;
#pragma unroll
    for (int t = 0; t < 8; ++t) {
      v16bf kb0 = frag_from_bf(krow0 + t * 32, hi);
      s0 = __builtin_amdgcn_wmma_f32_16x16x32_bf16(false, qa[t], false, kb0,
                                                   (short)0, s0, false, false);
      v16bf kb1 = frag_from_bf(krow1 + t * 32, hi);
      s1 = __builtin_amdgcn_wmma_f32_16x16x32_bf16(false, qa[t], false, kb1,
                                                   (short)0, s1, false, false);
    }

    // --- online softmax: rows live as (vgpr v, lane group) in D layout ---
    float mx[8];
#pragma unroll
    for (int v = 0; v < 8; ++v) mx[v] = fmaxf(s0[v], s1[v]);
#pragma unroll
    for (int m = 1; m <= 8; m <<= 1) {
#pragma unroll
      for (int v = 0; v < 8; ++v) mx[v] = fmaxf(mx[v], __shfl_xor(mx[v], m, 32));
    }
    float alpha[8];
#pragma unroll
    for (int v = 0; v < 8; ++v) {
      float mn = fmaxf(mrow[v], mx[v]);
      alpha[v] = __expf(mrow[v] - mn);
      mrow[v]  = mn;
    }
#pragma unroll
    for (int v = 0; v < 8; ++v) {
      s0[v] = __expf(s0[v] - mrow[v]);
      s1[v] = __expf(s1[v] - mrow[v]);
    }
    float rs[8];
#pragma unroll
    for (int v = 0; v < 8; ++v) rs[v] = s0[v] + s1[v];
#pragma unroll
    for (int m = 1; m <= 8; m <<= 1) {
#pragma unroll
      for (int v = 0; v < 8; ++v) rs[v] += __shfl_xor(rs[v], m, 32);
    }
#pragma unroll
    for (int v = 0; v < 8; ++v) lrow[v] = lrow[v] * alpha[v] + rs[v];
#pragma unroll
    for (int ct = 0; ct < 16; ++ct)
#pragma unroll
      for (int v = 0; v < 8; ++v) o[ct][v] *= alpha[v];

    // --- transpose P through per-wave LDS: D layout -> A layout ---
#pragma unroll
    for (int v = 0; v < 8; ++v) {
      Pls[w][v + 8 * hi][l16]      = (__bf16)s0[v];
      Pls[w][v + 8 * hi][16 + l16] = (__bf16)s1[v];
    }
    asm volatile("s_wait_dscnt 0" ::: "memory");   // wave-private RAW fence

    v16bf pa = frag_from_bf(&Pls[w][l16][0], hi);  // A: lane = query row, K = 32 keys

    // --- O += P @ V (Vt rows are contiguous key runs) ---
#pragma unroll
    for (int ct = 0; ct < 16; ++ct) {
      const __bf16* vrow = Vbase + (size_t)(ct * 16 + l16) * NSEQ + j0;
      v16bf vb = frag_from_bf(vrow, hi);
      o[ct] = __builtin_amdgcn_wmma_f32_16x16x32_bf16(false, pa, false, vb,
                                                      (short)0, o[ct], false, false);
    }
    asm volatile("s_wait_dscnt 0" ::: "memory");   // LDS reads done before next overwrite
  }

  float invl[8];
#pragma unroll
  for (int v = 0; v < 8; ++v) invl[v] = 1.f / lrow[v];
#pragma unroll
  for (int ct = 0; ct < 16; ++ct) {
    const int col = ct * 16 + l16;
#pragma unroll
    for (int v = 0; v < 8; ++v)
      out[(size_t)(r0 + v + 8 * hi) * HDIM + col] = o[ct][v] * invl[v];
  }
}

// ---------------------------------------------------------------------------
extern "C" void kernel_launch(void* const* d_in, const int* in_sizes, int n_in,
                              void* d_out, int out_size, void* d_ws, size_t ws_size,
                              hipStream_t stream) {
  (void)in_sizes; (void)n_in; (void)out_size; (void)ws_size;
  const float* x  = (const float*)d_in[0];
  const float* Wq = (const float*)d_in[1];
  const float* bq = (const float*)d_in[2];
  const float* Wk = (const float*)d_in[3];
  const float* bk = (const float*)d_in[4];
  const float* Wv = (const float*)d_in[5];
  const float* bv = (const float*)d_in[6];
  float* out = (float*)d_out;

  const size_t elems = (size_t)NB * NSEQ * HDIM;   // 4,194,304 per tensor
  __bf16* Qb = (__bf16*)d_ws;                      // 8 MB
  __bf16* Kb = Qb + elems;                         // 8 MB
  __bf16* Vt = Kb + elems;                         // 8 MB (transposed V)
  __bf16* xb = Vt + elems;                         // 8 MB (x in bf16)
  __bf16* Wb = xb + elems;                         // 384 KB (Wq|Wk|Wv in bf16)

  // One-shot fp32 -> bf16 conversions (x: 524288 vec8 chunks; W: 8192 each)
  cvt_bf16_kernel<<<2048, 256, 0, stream>>>(x,  xb, (int)(elems / 8));
  cvt_bf16_kernel<<<32,   256, 0, stream>>>(Wq, Wb + 0 * HDIM * HDIM, HDIM * HDIM / 8);
  cvt_bf16_kernel<<<32,   256, 0, stream>>>(Wk, Wb + 1 * HDIM * HDIM, HDIM * HDIM / 8);
  cvt_bf16_kernel<<<32,   256, 0, stream>>>(Wv, Wb + 2 * HDIM * HDIM, HDIM * HDIM / 8);

  // 49152 wave-tiles / 8 waves per 256-thread block
  qkv_proj_kernel<<<6144, 256, 0, stream>>>(xb, Wb, bq, bk, bv, Qb, Kb, Vt);
  // 1024 query-tile waves / 4 waves per 128-thread block
  flash_attn_kernel<<<256, 128, 0, stream>>>(Qb, Kb, Vt, out);
}
